// AttentionModule_27994596835863
// MI455X (gfx1250) — matrix operand order repaired and checked by reading, not compiled, not run on hardware
//
#include <hip/hip_runtime.h>

typedef __attribute__((ext_vector_type(16))) _Float16 v16h;
typedef __attribute__((ext_vector_type(8)))  _Float16 v8h;
typedef __attribute__((ext_vector_type(4)))  _Float16 h4;
typedef __attribute__((ext_vector_type(8)))  float    v8f;
typedef __attribute__((ext_vector_type(4)))  float    f32x4;

#define FD   256   // feature dim
#define SEQ  64    // both N (pedestrians) and T (time) are 64
#define LDA  264   // halves: row stride of 256-wide f16 LDS arrays (16B aligned, padded)
#define LDW  72    // halves: row stride of score/softmax buffer (64x64)

union F16x16 { v16h v; v8h h[2]; };

__device__ __forceinline__ v8f wmma(v16h a, v16h b, v8f c) {
  return __builtin_amdgcn_wmma_f32_16x16x32_f16(false, a, false, b, (short)0, c,
                                                false, false);
}

// A-fragment (16x32 f16): lane l -> row m0+(l&15); halves 0..15 = K in
// [k0+g*16,+16), g = l>>4.  Two 16B LDS loads.
__device__ __forceinline__ v16h frag_rowA(const _Float16* base, int ld, int row0, int k0) {
  const int l = threadIdx.x & 31;
  const _Float16* p = base + (row0 + (l & 15)) * ld + k0 + ((l >> 4) << 4);
  F16x16 r;
  r.h[0] = *(const v8h*)p;
  r.h[1] = *(const v8h*)(p + 8);
  return r.v;
}

// B-fragment (32x16 f16, K x N): lane l -> col n0+(l&15); halves 0..15 = K in
// [k0+g*16,+16), g = l>>4.  One contiguous 32B row read (two 16B loads).
__device__ __forceinline__ v16h frag_rowB(const _Float16* base, int ld, int col0, int k0) {
  const int l = threadIdx.x & 31;
  const _Float16* p = base + (col0 + (l & 15)) * ld + k0 + ((l >> 4) << 4);
  F16x16 r;
  r.h[0] = *(const v8h*)p;
  r.h[1] = *(const v8h*)(p + 8);
  return r.v;
}

// V^T lives unpadded (stride 64 halves) in the dead K buffer, XOR-swizzled in
// 8-half chunks so the row-parallel B reads are <=2-way bank conflicted.
__device__ __forceinline__ int vt_addr(int n, int cm) {   // halves offset of chunk
  return (n << 6) + ((cm ^ (n & 7)) << 3);
}

__device__ __forceinline__ v16h frag_B_vt(const _Float16* VT, int col0, int k0) {
  const int l = threadIdx.x & 31;
  const int n = col0 + (l & 15);
  const int cm0 = (k0 + ((l >> 4) << 4)) >> 3;
  F16x16 r;
  r.h[0] = *(const v8h*)(VT + vt_addr(n, cm0));
  r.h[1] = *(const v8h*)(VT + vt_addr(n, cm0 + 1));
  return r.v;
}

// Two output tiles (m0,n0) and (m0,n0+16) sharing every A fragment.
__device__ __forceinline__ void tile_gemm2(const _Float16* A, int lda,
                                           const _Float16* B, int ldb,
                                           int m0, int n0, int klen,
                                           v8f& acc0, v8f& acc1) {
  for (int k = 0; k < klen; k += 32) {
    v16h a  = frag_rowA(A, lda, m0, k);
    v16h b0 = frag_rowB(B, ldb, n0, k);
    v16h b1 = frag_rowB(B, ldb, n0 + 16, k);
    acc0 = wmma(a, b0, acc0);
    acc1 = wmma(a, b1, acc1);
  }
}

// ---- register double-buffered weight staging: plain struct of named members,
// moved by value only, so every chunk is an SSA value (no stack slot, no spill).
struct PF { uint4 a, b, c, d, e, f, g, h; };

template<int ROWS>
__device__ __forceinline__ PF wb_prefetch(const _Float16* __restrict__ W16, int tid) {
  const uint4* s = (const uint4*)W16;
  PF p;
  p.a = s[tid];
  p.b = s[tid + 256];
  p.c = s[tid + 512];
  p.d = s[tid + 768];
  if (ROWS == 64) {
    p.e = s[tid + 1024];
    p.f = s[tid + 1280];
    p.g = s[tid + 1536];
    p.h = s[tid + 1792];
  } else {
    p.e = p.a; p.f = p.a; p.g = p.a; p.h = p.a;   // dead, keeps struct defined
  }
  return p;
}

template<int ROWS>
__device__ __forceinline__ void wb_store(PF p, _Float16* dst, int tid) {
  const int r0 = tid >> 5;                  // rows advance by 8 per 256 chunks
  const int c0 = (tid & 31) << 3;
  *(uint4*)&dst[(r0     ) * LDA + c0] = p.a;
  *(uint4*)&dst[(r0 +  8) * LDA + c0] = p.b;
  *(uint4*)&dst[(r0 + 16) * LDA + c0] = p.c;
  *(uint4*)&dst[(r0 + 24) * LDA + c0] = p.d;
  if (ROWS == 64) {
    *(uint4*)&dst[(r0 + 32) * LDA + c0] = p.e;
    *(uint4*)&dst[(r0 + 40) * LDA + c0] = p.f;
    *(uint4*)&dst[(r0 + 48) * LDA + c0] = p.g;
    *(uint4*)&dst[(r0 + 56) * LDA + c0] = p.h;
  }
}

// One-time prep: convert the four 256x256 fp32 weight matrices to f16 (packed).
__global__ __launch_bounds__(256)
void cvt_weights(const float* __restrict__ Wq, const float* __restrict__ Wk,
                 const float* __restrict__ Wv, const float* __restrict__ Wo,
                 _Float16* __restrict__ out) {
  const int i = blockIdx.x * 256 + threadIdx.x;     // 65,536 float4 chunks
  const int m   = i >> 14;                          // 16,384 chunks per matrix
  const int off = (i & 16383) << 2;
  const float* src = (m & 2) ? ((m & 1) ? Wo : Wv) : ((m & 1) ? Wk : Wq);
  f32x4 v = *(const f32x4*)(src + off);
  h4 t = { (_Float16)v[0], (_Float16)v[1], (_Float16)v[2], (_Float16)v[3] };
  *(h4*)(out + (m << 16) + off) = t;
}

// One fused attention phase. Grid: 4096 blocks (one 64x256 slab each), 256 threads.
// Spatial: slab = x[b,t,:,:] (rows stride F).  Temporal: x[b,:,n,:] (stride N*F).
template<bool IN_F32, bool OUT_F32, bool TEMPORAL>
__global__ __launch_bounds__(256, 2)
void attn_phase(const void* __restrict__ in_,
                const _Float16* __restrict__ W16,
                const float* __restrict__ bq, const float* __restrict__ bk,
                const float* __restrict__ bv, const float* __restrict__ bo,
                void* __restrict__ out_) {
  __shared__ __align__(16) _Float16 Xs[SEQ * LDA];   // input slab (f16) + residual
  __shared__ __align__(16) _Float16 Qs[SEQ * LDA];   // Q, reused for att
  __shared__ __align__(16) _Float16 Ks[SEQ * LDA];   // K, reused for V^T (swizzled)
  __shared__ __align__(16) _Float16 Wst[SEQ * LDA];  // 64-row weight staging;
                                                     // front reused by scores/softmax
  const int tid  = threadIdx.x;
  const int wave = tid >> 5;
  const int lane = tid & 31;

  const _Float16* W16q = W16;
  const _Float16* W16k = W16 + (1 << 16);
  const _Float16* W16v = W16 + (2 << 16);
  const _Float16* W16o = W16 + (3 << 16);

  long base; int rs;
  if (TEMPORAL) {
    const int b = blockIdx.x >> 6, n = blockIdx.x & 63;
    base = (long)b * (SEQ * SEQ * FD) + (long)n * FD;
    rs = SEQ * FD;
  } else {
    base = (long)blockIdx.x * (SEQ * FD);
    rs = FD;
  }

  // Kick off the first weight block before touching the input slab so both
  // load streams are in flight together.
  PF pf = wb_prefetch<64>(W16q, tid);

  // ---- load input slab -> Xs (f16) ----
  if (IN_F32) {
    const float* in = (const float*)in_;
    for (int i = tid; i < SEQ * 64; i += 256) {
      const int r = i >> 6, c = (i & 63) << 2;
      f32x4 v = *(const f32x4*)(in + base + (long)r * rs + c);
      h4 t = { (_Float16)v[0], (_Float16)v[1], (_Float16)v[2], (_Float16)v[3] };
      *(h4*)&Xs[r * LDA + c] = t;
    }
  } else {
    const _Float16* in = (const _Float16*)in_;
    for (int i = tid; i < SEQ * 32; i += 256) {
      const int r = i >> 5, c = (i & 31) << 3;
      *(v8h*)&Xs[r * LDA + c] = *(const v8h*)(in + base + (long)r * rs + c);
    }
  }

  // 64x64 block tiling: wave -> m-tile (wave&3), n-half (wave>>2); 2 tiles/wave.
  const int m0  = (wave & 3) << 4;
  const int nh0 = (wave >> 2) << 5;
  const int mb  = m0 + ((lane >> 4) << 3);
  const int nl  = lane & 15;

  // ---- Q and K projections: Y = X @ W^T + b (64-row staged blocks,
  //      register double-buffered) ----
  for (int pass = 0; pass < 8; ++pass) {
    const int p = pass >> 2, nb = pass & 3;
    __syncthreads();                      // Wst readers of previous pass done
    wb_store<64>(pf, Wst, tid);           // implicit s_wait_loadcnt on pf
    __syncthreads();
    if (pass < 7) {                       // prefetch next block during GEMM
      const int q = pass + 1;
      pf = wb_prefetch<64>(((q >> 2) ? W16k : W16q) + (q & 3) * (64 * FD), tid);
    } else {
      pf = wb_prefetch<32>(W16v, tid);    // V block 0 hides behind scores/softmax
    }
    v8f a0 = {}, a1 = {};
    tile_gemm2(Xs, LDA, Wst, LDA, m0, nh0, FD, a0, a1);
    _Float16* D = p ? Ks : Qs;
    const float* bb = p ? bk : bq;
    const int nc = nb * 64 + nh0 + nl;
    const float b0 = bb[nc], b1 = bb[nc + 16];
#pragma unroll
    for (int r = 0; r < 8; ++r) {
      D[(mb + r) * LDA + nc]      = (_Float16)(a0[r] + b0);
      D[(mb + r) * LDA + nc + 16] = (_Float16)(a1[r] + b1);
    }
  }
  __syncthreads();

  // ---- scores = (Q @ K^T) * 1/sqrt(D) -> f16 in SW (front of Wst) ----
  _Float16* SW = Wst;
  {
    v8f a0 = {}, a1 = {};
    tile_gemm2(Qs, LDA, Ks, LDA, m0, nh0, FD, a0, a1);
    const int n = nh0 + nl;
#pragma unroll
    for (int r = 0; r < 8; ++r) {
      SW[(mb + r) * LDW + n]      = (_Float16)(a0[r] * 0.125f);   // 1/sqrt(64)
      SW[(mb + r) * LDW + n + 16] = (_Float16)(a1[r] * 0.125f);
    }
  }
  __syncthreads();

  // ---- softmax over rows (thread-per-row, f32 math) ----
  if (tid < SEQ) {
    _Float16* row = &SW[tid * LDW];
    float mx = -1e30f;
    for (int j = 0; j < SEQ; ++j) mx = fmaxf(mx, (float)row[j]);
    float s = 0.f;
    for (int j = 0; j < SEQ; ++j) {
      float e = __expf((float)row[j] - mx);
      s += e;
      row[j] = (_Float16)e;
    }
    const float inv = 1.f / s;
    for (int j = 0; j < SEQ; ++j) row[j] = (_Float16)((float)row[j] * inv);
  }
  __syncthreads();

  // ---- V projection -> V^T (swizzled) into dead K buffer; staging area sits
  //      past the live 9 KB softmax region inside Wst. ----
  _Float16* VTs = Ks;
  _Float16* Vst = Wst + SEQ * LDW;
  for (int nb = 0; nb < 8; ++nb) {
    __syncthreads();
    wb_store<32>(pf, Vst, tid);
    __syncthreads();
    if (nb < 7) pf = wb_prefetch<32>(W16v + (nb + 1) * (32 * FD), tid);
    else        pf = wb_prefetch<64>(W16o, tid);  // O block 0 hides behind w@V
    const int vm0 = (wave >> 1) << 4;             // 4m x 2n tiles, 1 per wave
    const int vn0 = (wave & 1) << 4;
    v8f acc = {};
    for (int k = 0; k < FD; k += 32) {
      v16h a = frag_rowA(Xs, LDA, vm0, k);
      v16h b = frag_rowB(Vst, LDA, vn0, k);
      acc = wmma(a, b, acc);
    }
    const int ncol = nb * 32 + vn0 + nl;          // feature col = V^T row
    const float bv2 = bv[ncol];
    const int vmb = vm0 + ((lane >> 4) << 3);     // multiple of 8
    h4 lo = { (_Float16)(acc[0] + bv2), (_Float16)(acc[1] + bv2),
              (_Float16)(acc[2] + bv2), (_Float16)(acc[3] + bv2) };
    h4 hi = { (_Float16)(acc[4] + bv2), (_Float16)(acc[5] + bv2),
              (_Float16)(acc[6] + bv2), (_Float16)(acc[7] + bv2) };
    _Float16* d = VTs + vt_addr(ncol, vmb >> 3);  // 8 contiguous halves
    *(h4*)d = lo;
    *(h4*)(d + 4) = hi;
  }
  __syncthreads();

  // ---- att = w @ V (contract 64): two passes of 4 n-tiles per wave, each
  //      sharing the A fragment (keeps peak VGPR pressure low). ----
  _Float16* Att = Qs;                              // Q is dead
#pragma unroll
  for (int half = 0; half < 2; ++half) {
    v8f c0 = {}, c1 = {}, c2 = {}, c3 = {};
    const int nb0 = ((wave >> 2) << 3) + (half << 2);  // 4 of 16 n-tiles
    for (int k = 0; k < SEQ; k += 32) {
      v16h a = frag_rowA(SW, LDW, m0, k);
      c0 = wmma(a, frag_B_vt(VTs, (nb0    ) << 4, k), c0);
      c1 = wmma(a, frag_B_vt(VTs, (nb0 + 1) << 4, k), c1);
      c2 = wmma(a, frag_B_vt(VTs, (nb0 + 2) << 4, k), c2);
      c3 = wmma(a, frag_B_vt(VTs, (nb0 + 3) << 4, k), c3);
    }
#pragma unroll
    for (int r = 0; r < 8; ++r) {
      _Float16* dr = &Att[(mb + r) * LDA + nl];
      dr[(nb0    ) << 4] = (_Float16)c0[r];
      dr[(nb0 + 1) << 4] = (_Float16)c1[r];
      dr[(nb0 + 2) << 4] = (_Float16)c2[r];
      dr[(nb0 + 3) << 4] = (_Float16)c3[r];
    }
  }

  // ---- out = att @ Wo^T + bo + residual -> global ----
  for (int nb = 0; nb < 4; ++nb) {
    __syncthreads();                               // att (SW/Wst readers) done
    wb_store<64>(pf, Wst, tid);                    // clobbers SW (dead)
    __syncthreads();
    if (nb < 3) pf = wb_prefetch<64>(W16o + (nb + 1) * (64 * FD), tid);
    v8f a0 = {}, a1 = {};
    tile_gemm2(Att, LDA, Wst, LDA, m0, nh0, FD, a0, a1);
    const int nc = nb * 64 + nh0 + nl;
    const float b0 = bo[nc], b1 = bo[nc + 16];
#pragma unroll
    for (int r = 0; r < 8; ++r) {
      const float v0 = a0[r] + b0 + (float)Xs[(mb + r) * LDA + nc];
      const float v1 = a1[r] + b1 + (float)Xs[(mb + r) * LDA + nc + 16];
      const long i0 = base + (long)(mb + r) * rs + nc;
      if (OUT_F32) {
        ((float*)out_)[i0]      = v0;
        ((float*)out_)[i0 + 16] = v1;
      } else {
        ((_Float16*)out_)[i0]      = (_Float16)v0;
        ((_Float16*)out_)[i0 + 16] = (_Float16)v1;
      }
    }
  }
}

extern "C" void kernel_launch(void* const* d_in, const int* in_sizes, int n_in,
                              void* d_out, int out_size, void* d_ws, size_t ws_size,
                              hipStream_t stream) {
  const float* x  = (const float*)d_in[0];
  const float* Wq = (const float*)d_in[1];
  const float* bq = (const float*)d_in[2];
  const float* Wk = (const float*)d_in[3];
  const float* bk = (const float*)d_in[4];
  const float* Wv = (const float*)d_in[5];
  const float* bv = (const float*)d_in[6];
  const float* Wo = (const float*)d_in[7];
  const float* bo = (const float*)d_in[8];

  // Workspace layout: xs (f16 intermediate, 128 MiB) | W16 (4 matrices, 512 KiB)
  _Float16* xs  = (_Float16*)d_ws;
  _Float16* W16 = (_Float16*)((char*)d_ws + (size_t)64 * 64 * 64 * 256 * 2);

  // One-time weight conversion fp32 -> f16 (re-run every call: deterministic).
  cvt_weights<<<256, 256, 0, stream>>>(Wq, Wk, Wv, Wo, W16);

  dim3 grid(64 * 64), block(256);
  // Phase 1: spatial attention, fp32 in -> f16 xs
  attn_phase<true,  false, false><<<grid, block, 0, stream>>>(
      x,  W16, bq, bk, bv, bo, xs);
  // Phase 2: temporal attention, f16 xs -> fp32 out
  attn_phase<false, true,  true ><<<grid, block, 0, stream>>>(
      xs, W16, bq, bk, bv, bo, d_out);
}
